// NSE_42984032699092
// MI455X (gfx1250) — compile-verified, matrix operand order checked by reading, not run on hardware
//
#include <hip/hip_runtime.h>
#include <stdint.h>

// Problem constants (match reference)
#define T_STEPS 32
#define BB      128    // batch
#define LL      1024   // memory slots
#define DD      256    // hidden
#define FOURD   1024   // 4*D

// ---------------- WMMA / SGPR-tuple types ----------------
typedef __attribute__((ext_vector_type(16))) __bf16 v16bf;
typedef __attribute__((ext_vector_type(8)))  float  v8f;
typedef uint32_t v4u __attribute__((ext_vector_type(4)));
typedef uint32_t v8u __attribute__((ext_vector_type(8)));

union FragBF {
  uint32_t u[8];
  uint4    q[2];
  v16bf    v;
};

__device__ __forceinline__ uint16_t f2bf(float f) {
  uint32_t u = __float_as_uint(f);
  u += 0x7FFFu + ((u >> 16) & 1u);          // round-to-nearest-even
  return (uint16_t)(u >> 16);
}

__device__ __forceinline__ float sigmoidf_(float x) {
  return 1.0f / (1.0f + __expf(-x));
}

// ---------------- f32 -> bf16 convert ----------------
__global__ void cvt_f32_to_bf16(const float* __restrict__ src,
                                uint16_t* __restrict__ dst, int n) {
  int i = blockIdx.x * blockDim.x + threadIdx.x;
  if (i < n) dst[i] = f2bf(src[i]);
}

// ---------------- dual-source WMMA GEMM ----------------
// C[M,N] = A1[M,K]*W1[N,K]^T (+ A2[M,K]*W2[N,K]^T) + bias1[N] + bias2[N]
// bf16 inputs, f32 accumulate via v_wmma_f32_16x16x32_bf16.
// Block = 256 threads = 8 waves; each wave owns one 16x16 output tile;
// a block covers a 16(M) x 128(N) strip. Grid = (N/128, M/16).
__global__ void __launch_bounds__(256)
gemm_dual_bf16_wmma(const uint16_t* __restrict__ A1, const uint16_t* __restrict__ W1,
                    const uint16_t* __restrict__ A2, const uint16_t* __restrict__ W2,
                    const float* __restrict__ bias1, const float* __restrict__ bias2,
                    float* __restrict__ C, int N, int K) {
  const int wave = threadIdx.x >> 5;
  const int lane = threadIdx.x & 31;
  const int row0 = blockIdx.y << 4;
  const int col0 = (blockIdx.x << 7) + (wave << 4);
  const int r  = lane & 15;
  const int kg = lane >> 4;

  v8f acc = {};
  {
    const uint16_t* ap = A1 + (size_t)(row0 + r) * K + kg * 8;
    const uint16_t* bp = W1 + (size_t)(col0 + r) * K + kg * 8;
    for (int k0 = 0; k0 < K; k0 += 32) {
      FragBF a, b;
      a.q[0] = *(const uint4*)(ap + k0);
      a.q[1] = *(const uint4*)(ap + k0 + 16);
      b.q[0] = *(const uint4*)(bp + k0);
      b.q[1] = *(const uint4*)(bp + k0 + 16);
      acc = __builtin_amdgcn_wmma_f32_16x16x32_bf16(false, a.v, false, b.v,
                                                    (short)0, acc, false, false);
    }
  }
  if (A2 != nullptr) {
    const uint16_t* ap = A2 + (size_t)(row0 + r) * K + kg * 8;
    const uint16_t* bp = W2 + (size_t)(col0 + r) * K + kg * 8;
    for (int k0 = 0; k0 < K; k0 += 32) {
      FragBF a, b;
      a.q[0] = *(const uint4*)(ap + k0);
      a.q[1] = *(const uint4*)(ap + k0 + 16);
      b.q[0] = *(const uint4*)(bp + k0);
      b.q[1] = *(const uint4*)(bp + k0 + 16);
      acc = __builtin_amdgcn_wmma_f32_16x16x32_bf16(false, a.v, false, b.v,
                                                    (short)0, acc, false, false);
    }
  }

  const int n = col0 + r;
  float bias = 0.0f;
  if (bias1) bias += bias1[n];
  if (bias2) bias += bias2[n];
  const int mhalf = kg * 8;
#pragma unroll
  for (int i = 0; i < 8; ++i)
    C[(size_t)(row0 + mhalf + i) * N + n] = acc[i] + bias;
}

// ---------------- LSTM pointwise ----------------
__global__ void __launch_bounds__(256)
lstm_pointwise(const float* __restrict__ G, float* __restrict__ c,
               float* __restrict__ h, uint16_t* __restrict__ hbf,
               uint16_t* __restrict__ catbf, float* __restrict__ out_slot) {
  int idx = blockIdx.x * 256 + threadIdx.x;       // [0, B*D)
  int b = idx >> 8;                               // D == 256
  int d = idx & 255;
  const float* g = G + (size_t)b * FOURD;
  float ig = sigmoidf_(g[d]);
  float fg = sigmoidf_(g[DD + d]);
  float gg = tanhf(g[2 * DD + d]);
  float og = sigmoidf_(g[3 * DD + d]);
  float c2 = fg * c[idx] + ig * gg;
  float hv = og * tanhf(c2);
  c[idx] = c2;
  h[idx] = hv;
  hbf[idx] = f2bf(hv);
  if (catbf)    catbf[(size_t)b * (2 * DD) + d] = f2bf(hv);  // first half of [hr|m]
  if (out_slot) out_slot[idx] = hv;
}

// ---------------- attention logits via TDM ----------------
// s[b,l] = hr[b,:].M[b,l,:] (masked -inf). One block per (b, 32 l-rows).
// Wave 0 issues a single tensor_load_to_lds DMA of 32 contiguous M rows
// (32 KB, described as a 1-D tile of 8192 f32), waits on TENSORcnt, then
// all 8 waves compute 4 rows each from LDS with shuffle reductions.
__global__ void __launch_bounds__(256)
attn_logits_kernel(const float* __restrict__ hr, const float* __restrict__ M,
                   const unsigned char* __restrict__ mask, float* __restrict__ s) {
  __shared__ __align__(16) float mtile[32 * DD];   // 32 KB
  int b = blockIdx.y;
  int l0 = blockIdx.x * 32;
  int wave = threadIdx.x >> 5, lane = threadIdx.x & 31;

  if (wave == 0) {
    uint64_t gaddr = (uint64_t)(uintptr_t)(M + ((size_t)b * LL + l0) * DD);
    uint32_t laddr = (uint32_t)(uintptr_t)&mtile[0];   // low 32 bits = LDS offset
    const uint32_t elems = 32 * DD;                    // 8192 elements of 4 B

    v4u g0;                                  // D# group 0
    g0[0] = 1u;                              // count=1 (valid user descriptor)
    g0[1] = laddr;                           // lds_addr
    g0[2] = (uint32_t)gaddr;                 // global_addr[31:0]
    g0[3] = (uint32_t)((gaddr >> 32) & 0x01FFFFFFu) | (2u << 30); // ga[56:32]|type=2

    v8u g1;                                  // D# group 1
    g1[0] = 2u << 16;                        // data_size=4B; mask/flags=0
    g1[1] = (elems & 0xFFFFu) << 16;         // tensor_dim0[15:0]
    g1[2] = (elems >> 16) & 0xFFFFu;         // tensor_dim0[31:16]; tensor_dim1=0
    g1[3] = (elems & 0xFFFFu) << 16;         // tile_dim0 = 8192
    g1[4] = 0u;                              // tile_dim1/2 unused (1-D tile)
    g1[5] = elems;                           // tensor_dim0_stride[31:0]
    g1[6] = 0u;
    g1[7] = 0u;

    v4u g2 = {0u, 0u, 0u, 0u};               // groups 2/3 present but unused
    v4u g3 = {0u, 0u, 0u, 0u};

    asm volatile("tensor_load_to_lds %0, %1, %2, %3"
                 :: "s"(g0), "s"(g1), "s"(g2), "s"(g3) : "memory");
    __builtin_amdgcn_s_wait_tensorcnt(0);
  }
  __syncthreads();

  const float4* hp = (const float4*)(hr + (size_t)b * DD) + lane * 2;
  float4 h0 = hp[0], h1 = hp[1];
#pragma unroll
  for (int j = 0; j < 4; ++j) {
    int row = wave * 4 + j;
    const float4* mp = (const float4*)(mtile + row * DD) + lane * 2;
    float4 m0 = mp[0], m1 = mp[1];
    float dot = h0.x * m0.x + h0.y * m0.y + h0.z * m0.z + h0.w * m0.w +
                h1.x * m1.x + h1.y * m1.y + h1.z * m1.z + h1.w * m1.w;
#pragma unroll
    for (int off = 16; off > 0; off >>= 1) dot += __shfl_xor(dot, off, 32);
    if (lane == 0) {
      int l = l0 + row;
      s[(size_t)b * LL + l] =
          mask[(size_t)b * LL + l] ? -__builtin_inff() : dot;
    }
  }
}

// ---------------- softmax over L (in place) ----------------
__global__ void __launch_bounds__(256)
softmax_z_kernel(float* __restrict__ z) {
  int b = blockIdx.x, tid = threadIdx.x;
  __shared__ float red[256];
  float vals[4], vmax = -__builtin_inff();
#pragma unroll
  for (int i = 0; i < 4; ++i) {
    vals[i] = z[(size_t)b * LL + tid + i * 256];
    vmax = fmaxf(vmax, vals[i]);
  }
  red[tid] = vmax; __syncthreads();
  for (int st = 128; st > 0; st >>= 1) {
    if (tid < st) red[tid] = fmaxf(red[tid], red[tid + st]);
    __syncthreads();
  }
  float mx = red[0]; __syncthreads();
  float sum = 0.0f;
#pragma unroll
  for (int i = 0; i < 4; ++i) { vals[i] = __expf(vals[i] - mx); sum += vals[i]; }
  red[tid] = sum; __syncthreads();
  for (int st = 128; st > 0; st >>= 1) {
    if (tid < st) red[tid] += red[tid + st];
    __syncthreads();
  }
  float inv = 1.0f / red[0];
#pragma unroll
  for (int i = 0; i < 4; ++i) z[(size_t)b * LL + tid + i * 256] = vals[i] * inv;
}

// ---------------- m = z . M  (async-LDS double-buffered, deterministic) ----------------
// Block (chunk c, batch b) accumulates 256 L-rows in 16-row (16 KB) chunks
// streamed through LDS with global_load_async_to_lds_b128; one chunk kept in
// flight (s_wait_asynccnt 0x4). Each thread owns one output column d, so the
// per-row LDS reads across the block are conflict-free and the global stream
// is pure 128-bit async traffic.
__global__ void __launch_bounds__(256)
attn_m_partial(const float* __restrict__ z, const float* __restrict__ M,
               float* __restrict__ partial) {
  __shared__ __align__(16) float mbuf[2][16 * DD];   // 2 x 16 KB
  __shared__ float zs[256];
  int c = blockIdx.x, b = blockIdx.y, tid = threadIdx.x;
  zs[tid] = z[(size_t)b * LL + c * 256 + tid];

  const char* gbase = (const char*)(M + ((size_t)b * LL + c * 256) * DD);
  uint32_t lbuf0 = (uint32_t)(uintptr_t)&mbuf[0][0] + tid * 64u;
  uint32_t lbuf1 = (uint32_t)(uintptr_t)&mbuf[1][0] + tid * 64u;

  // prologue: chunk 0 -> buffer 0 (64 B per thread = 4 x b128; the offset:
  // immediate applies to BOTH the LDS and global address, so one base pair
  // describes a flat 64 B copy)
  {
    uint64_t g = (uint64_t)(uintptr_t)gbase + (uint64_t)tid * 64u;
    asm volatile(
        "global_load_async_to_lds_b128 %0, %1, off\n\t"
        "global_load_async_to_lds_b128 %0, %1, off offset:16\n\t"
        "global_load_async_to_lds_b128 %0, %1, off offset:32\n\t"
        "global_load_async_to_lds_b128 %0, %1, off offset:48"
        :: "v"(lbuf0), "v"(g) : "memory");
  }

  float acc = 0.0f;
  for (int ch = 0; ch < 16; ++ch) {
    if (ch + 1 < 16) {
      uint64_t g = (uint64_t)(uintptr_t)gbase +
                   (uint64_t)(ch + 1) * 16384u + (uint64_t)tid * 64u;
      uint32_t l = ((ch + 1) & 1) ? lbuf1 : lbuf0;
      asm volatile(
          "global_load_async_to_lds_b128 %0, %1, off\n\t"
          "global_load_async_to_lds_b128 %0, %1, off offset:16\n\t"
          "global_load_async_to_lds_b128 %0, %1, off offset:32\n\t"
          "global_load_async_to_lds_b128 %0, %1, off offset:48"
          :: "v"(l), "v"(g) : "memory");
      asm volatile("s_wait_asynccnt 0x4" ::: "memory");  // chunk ch complete
    } else {
      asm volatile("s_wait_asynccnt 0x0" ::: "memory");
    }
    __syncthreads();                     // publish chunk ch across waves
    const float* mb = mbuf[ch & 1];
    const float* zrow = zs + ch * 16;
#pragma unroll
    for (int i = 0; i < 16; ++i)
      acc += zrow[i] * mb[i * DD + tid];
    __syncthreads();                     // WAR: done reading before next fill
  }
  partial[((size_t)c * BB + b) * DD + tid] = acc;
}

__global__ void __launch_bounds__(256)
attn_m_reduce(const float* __restrict__ partial, uint16_t* __restrict__ catbf) {
  int idx = blockIdx.x * 256 + threadIdx.x;     // [0, B*D)
  int b = idx >> 8, d = idx & 255;
  float m = partial[idx] + partial[(size_t)BB * DD + idx] +
            partial[(size_t)2 * BB * DD + idx] + partial[(size_t)3 * BB * DD + idx];
  catbf[(size_t)b * (2 * DD) + DD + d] = f2bf(m);  // second half of [hr|m]
}

// ---------------- softmax over D for comp (f32 logits -> bf16) ----------------
__global__ void __launch_bounds__(256)
softmax_comp_kernel(const float* __restrict__ G, uint16_t* __restrict__ compbf) {
  int b = blockIdx.x, d = threadIdx.x;          // 256 threads == D
  __shared__ float red[256];
  float v = G[(size_t)b * DD + d];
  red[d] = v; __syncthreads();
  for (int st = 128; st > 0; st >>= 1) {
    if (d < st) red[d] = fmaxf(red[d], red[d + st]);
    __syncthreads();
  }
  float mx = red[0]; __syncthreads();
  float e = __expf(v - mx);
  red[d] = e; __syncthreads();
  for (int st = 128; st > 0; st >>= 1) {
    if (d < st) red[d] += red[d + st];
    __syncthreads();
  }
  compbf[(size_t)b * DD + d] = f2bf(e / red[0]);
}

// ---------------- memory rewrite: M = (1-z) + hw (x) z  (write-only) ----------------
__global__ void __launch_bounds__(256)
rewrite_kernel(float* __restrict__ M, const float* __restrict__ z,
               const float* __restrict__ hw) {
  int b = blockIdx.y, l0 = blockIdx.x * 8, d = threadIdx.x;
  float h = hw[(size_t)b * DD + d];
#pragma unroll
  for (int i = 0; i < 8; ++i) {
    float zz = z[(size_t)b * LL + l0 + i];
    M[((size_t)b * LL + l0 + i) * DD + d] = (1.0f - zz) + h * zz;
  }
}

// ==================================================================
extern "C" void kernel_launch(void* const* d_in, const int* in_sizes, int n_in,
                              void* d_out, int out_size, void* d_ws, size_t ws_size,
                              hipStream_t stream) {
  const float* emb   = (const float*)d_in[0];
  const float* hr0   = (const float*)d_in[1];
  const float* cr0   = (const float*)d_in[2];
  const float* hw0   = (const float*)d_in[3];
  const float* cw0   = (const float*)d_in[4];
  const float* M0    = (const float*)d_in[5];
  const unsigned char* mask = (const unsigned char*)d_in[6];
  const float* Wih_r = (const float*)d_in[7];
  const float* Whh_r = (const float*)d_in[8];
  const float* bih_r = (const float*)d_in[9];
  const float* bhh_r = (const float*)d_in[10];
  const float* Wc    = (const float*)d_in[11];
  const float* bc    = (const float*)d_in[12];
  const float* Wih_w = (const float*)d_in[13];
  const float* Whh_w = (const float*)d_in[14];
  const float* bih_w = (const float*)d_in[15];
  const float* bhh_w = (const float*)d_in[16];

  // Output layout: outputs[T,B,D] | hr | cr | hw | cw | M  (states live here)
  float* out_outputs = (float*)d_out;
  float* hr = out_outputs + (size_t)T_STEPS * BB * DD;
  float* cr = hr + (size_t)BB * DD;
  float* hw = cr + (size_t)BB * DD;
  float* cw = hw + (size_t)BB * DD;
  float* Ms = cw + (size_t)BB * DD;

  // Workspace layout (16B-aligned, ~4.3 MB total)
  char* ws = (char*)d_ws;
  uint16_t* wihr_bf = (uint16_t*)(ws + 0);         // 1024*256 bf16
  uint16_t* whhr_bf = (uint16_t*)(ws + 524288);
  uint16_t* wihw_bf = (uint16_t*)(ws + 1048576);
  uint16_t* whhw_bf = (uint16_t*)(ws + 1572864);
  uint16_t* wc_bf   = (uint16_t*)(ws + 2097152);   // 256*512 bf16
  float*    G       = (float*)   (ws + 2359296);   // 128*1024 f32
  uint16_t* xbf     = (uint16_t*)(ws + 2883584);   // 128*256 bf16
  uint16_t* hrbf    = (uint16_t*)(ws + 2949120);
  uint16_t* hwbf    = (uint16_t*)(ws + 3014656);
  uint16_t* compbf  = (uint16_t*)(ws + 3080192);
  uint16_t* catbf   = (uint16_t*)(ws + 3145728);   // 128*512 bf16
  float*    zbuf    = (float*)   (ws + 3276800);   // 128*1024 f32 (logits -> z)
  float*    partial = (float*)   (ws + 3801088);   // 4*128*256 f32

  // ---- state init (d2d, capture-safe) ----
  hipMemcpyAsync(hr, hr0, (size_t)BB * DD * 4, hipMemcpyDeviceToDevice, stream);
  hipMemcpyAsync(cr, cr0, (size_t)BB * DD * 4, hipMemcpyDeviceToDevice, stream);
  hipMemcpyAsync(hw, hw0, (size_t)BB * DD * 4, hipMemcpyDeviceToDevice, stream);
  hipMemcpyAsync(cw, cw0, (size_t)BB * DD * 4, hipMemcpyDeviceToDevice, stream);
  hipMemcpyAsync(Ms, M0, (size_t)BB * LL * DD * 4, hipMemcpyDeviceToDevice, stream);

  // ---- weight / initial-state bf16 conversion ----
  cvt_f32_to_bf16<<<FOURD * DD / 256, 256, 0, stream>>>(Wih_r, wihr_bf, FOURD * DD);
  cvt_f32_to_bf16<<<FOURD * DD / 256, 256, 0, stream>>>(Whh_r, whhr_bf, FOURD * DD);
  cvt_f32_to_bf16<<<FOURD * DD / 256, 256, 0, stream>>>(Wih_w, wihw_bf, FOURD * DD);
  cvt_f32_to_bf16<<<FOURD * DD / 256, 256, 0, stream>>>(Whh_w, whhw_bf, FOURD * DD);
  cvt_f32_to_bf16<<<DD * 2 * DD / 256, 256, 0, stream>>>(Wc, wc_bf, DD * 2 * DD);
  cvt_f32_to_bf16<<<BB * DD / 256, 256, 0, stream>>>(hr0, hrbf, BB * DD);
  cvt_f32_to_bf16<<<BB * DD / 256, 256, 0, stream>>>(hw0, hwbf, BB * DD);

  const dim3 blk(256);
  for (int t = 0; t < T_STEPS; ++t) {
    // x_t -> bf16
    cvt_f32_to_bf16<<<BB * DD / 256, blk, 0, stream>>>(emb + (size_t)t * BB * DD,
                                                       xbf, BB * DD);
    // read LSTM gates: G = x.WihT + hr.WhhT + b
    gemm_dual_bf16_wmma<<<dim3(FOURD / 128, BB / 16), blk, 0, stream>>>(
        xbf, wihr_bf, hrbf, whhr_bf, bih_r, bhh_r, G, FOURD, DD);
    lstm_pointwise<<<BB * DD / 256, blk, 0, stream>>>(G, cr, hr, hrbf, catbf, nullptr);

    // attention: TDM-staged logits -> softmax -> async-staged m (into cat[:,256:])
    attn_logits_kernel<<<dim3(LL / 32, BB), blk, 0, stream>>>(hr, Ms, mask, zbuf);
    softmax_z_kernel<<<BB, blk, 0, stream>>>(zbuf);
    attn_m_partial<<<dim3(4, BB), blk, 0, stream>>>(zbuf, Ms, partial);
    attn_m_reduce<<<BB * DD / 256, blk, 0, stream>>>(partial, catbf);

    // comp = softmax([hr|m].WcT + bc)
    gemm_dual_bf16_wmma<<<dim3(DD / 128, BB / 16), blk, 0, stream>>>(
        catbf, wc_bf, nullptr, nullptr, bc, nullptr, G, DD, 2 * DD);
    softmax_comp_kernel<<<BB, blk, 0, stream>>>(G, compbf);

    // write LSTM
    gemm_dual_bf16_wmma<<<dim3(FOURD / 128, BB / 16), blk, 0, stream>>>(
        compbf, wihw_bf, hwbf, whhw_bf, bih_w, bhh_w, G, FOURD, DD);
    lstm_pointwise<<<BB * DD / 256, blk, 0, stream>>>(
        G, cw, hw, hwbf, nullptr, out_outputs + (size_t)t * BB * DD);

    // memory rewrite (full overwrite, no read of old M)
    rewrite_kernel<<<dim3(LL / 8, BB), blk, 0, stream>>>(Ms, zbuf, hw);
  }
}
